// EnergyConditionedAtomAttention_77627238908548
// MI455X (gfx1250) — matrix operand order repaired and checked by reading, not compiled, not run on hardware
//
#include <hip/hip_runtime.h>
#include <hip/hip_bf16.h>

typedef __attribute__((ext_vector_type(16))) _Float16 v16h;
typedef __attribute__((ext_vector_type(8)))  _Float16 v8h;
typedef __attribute__((ext_vector_type(8)))  float    v8f;
typedef __attribute__((ext_vector_type(4)))  int      v4i;

namespace {
constexpr int B_      = 16;
constexpr int NATOM   = 64;
constexpr int AD      = 128;
constexpr int ED      = 64;
constexpr int NE      = 256;
constexpr int LATENT  = 256;
constexpr int NH      = 8;
constexpr int HD      = 32;
constexpr int KQ      = 192;   // AD + ED, multiple of 32
constexpr int KATOM   = 224;   // 196 padded up to multiple of 32
constexpr int KATOM_R = 196;   // real feature width
constexpr float RCUT  = 6.0f;
}

// ---------------------------------------------------------------------------
// WMMA fragment loaders (wave32 layouts per CDNA5 ISA 7.12.2).
// A (16x32 f16, MxK): lane L holds row m=L&15; VGPR0-3 = K {0..7}(+8 if L>=16),
// VGPR4-7 = K {16..23}(+8 if L>=16).
// B (32x16 f16, KxN) staged transposed as Bt[n][k]: lane L holds col n=L&15;
// halves 0..7 = K {0..7}(+8 if L>=16), halves 8..15 = K {16..23}(+8 if L>=16).
// ---------------------------------------------------------------------------
__device__ __forceinline__ v16h frag_a_lds(const _Float16* base, int ld) {
  const int lane = threadIdx.x & 31;
  const _Float16* p = base + (lane & 15) * ld + ((lane & 16) ? 8 : 0);
  v8h lo = *(const v8h*)(p);
  v8h hi = *(const v8h*)(p + 16);
  v16h r;
#pragma unroll
  for (int i = 0; i < 8; ++i) { r[i] = lo[i]; r[i + 8] = hi[i]; }
  return r;
}

__device__ __forceinline__ v16h frag_b_lds(const _Float16* base, int ld) {
  const int lane = threadIdx.x & 31;
  const _Float16* p = base + (lane & 15) * ld + ((lane & 16) ? 8 : 0);
  v8h lo = *(const v8h*)(p);
  v8h hi = *(const v8h*)(p + 16);
  v16h r;
#pragma unroll
  for (int i = 0; i < 8; ++i) { r[i] = lo[i]; r[i + 8] = hi[i]; }
  return r;
}

__device__ __forceinline__ v8f wmma_f16(v16h a, v16h b, v8f c) {
  return __builtin_amdgcn_wmma_f32_16x16x32_f16(false, a, false, b,
                                                (short)0, c, false, false);
}

__device__ __forceinline__ float silu_f(float x) {
  return x / (1.0f + __expf(-x));
}

// Wave-relative LDS byte offset of a generic pointer into __shared__ memory.
typedef __attribute__((address_space(3))) void lds_void;
__device__ __forceinline__ unsigned lds_off(const void* p) {
  return (unsigned)(unsigned long long)(lds_void*)p;
}

// Async 16B global -> LDS copy (CDNA5 GLOBAL_LOAD_ASYNC_TO_LDS_B128, tracked
// by ASYNCcnt; INST_OFFSET is added to BOTH the global and LDS addresses).
__device__ __forceinline__ void async_g2l_b128(unsigned ldsByteOff,
                                               const void* gaddr) {
  asm volatile("global_load_async_to_lds_b128 %0, %1, off"
               :
               : "v"(ldsByteOff), "v"(gaddr)
               : "memory");
}
__device__ __forceinline__ void async_g2l_b128_o16(unsigned ldsByteOff,
                                                   const void* gaddr) {
  asm volatile("global_load_async_to_lds_b128 %0, %1, off offset:16"
               :
               : "v"(ldsByteOff), "v"(gaddr)
               : "memory");
}
__device__ __forceinline__ void wait_async_le4() {
  asm volatile("s_wait_asynccnt 0x4" ::: "memory");
}
__device__ __forceinline__ void wait_async_0() {
  asm volatile("s_wait_asynccnt 0x0" ::: "memory");
}

// ---------------------------------------------------------------------------
// Prep kernels
// ---------------------------------------------------------------------------
__global__ __launch_bounds__(256) void wt_cvt_kernel(
    const float* __restrict__ src, _Float16* __restrict__ dst,
    int K, int Kpad, int N) {
  int idx = blockIdx.x * blockDim.x + threadIdx.x;  // over N*Kpad
  int n = idx / Kpad;
  int kp = idx - n * Kpad;
  if (n < N)
    dst[idx] = (kp < K) ? (_Float16)src[(size_t)kp * N + n] : (_Float16)0.f;
}

__global__ __launch_bounds__(256) void build_qin_kernel(
    const float* __restrict__ h, const float* __restrict__ e_feat,
    const int* __restrict__ absIdx, _Float16* __restrict__ qin) {
  int idx = blockIdx.x * blockDim.x + threadIdx.x;  // over B*NE*KQ
  int row = idx / KQ, c = idx - row * KQ;
  int b = row / NE, e = row - b * NE;
  int ai = absIdx[0];
  float v = (c < AD) ? h[((size_t)b * NATOM + ai) * AD + c]
                     : e_feat[(size_t)e * ED + (c - AD)];
  qin[idx] = (_Float16)v;
}

__global__ __launch_bounds__(256) void build_atom_kernel(
    const float* __restrict__ h, const int* __restrict__ z,
    const float* __restrict__ pos, const unsigned char* __restrict__ mask,
    const float* __restrict__ z_emb, const int* __restrict__ absIdx,
    _Float16* __restrict__ atom, float* __restrict__ logcut,
    int* __restrict__ validArr) {
  int bn = blockIdx.x;                 // 0 .. B*NATOM-1
  int b = bn / NATOM, n = bn - b * NATOM;
  int ai = absIdx[0];
  float px = pos[(size_t)bn * 3 + 0] - pos[((size_t)b * NATOM + ai) * 3 + 0];
  float py = pos[(size_t)bn * 3 + 1] - pos[((size_t)b * NATOM + ai) * 3 + 1];
  float pz = pos[(size_t)bn * 3 + 2] - pos[((size_t)b * NATOM + ai) * 3 + 2];
  float r = sqrtf(px * px + py * py + pz * pz);
  float rs = fmaxf(r, 1e-8f);
  int c = threadIdx.x;
  if (c < KATOM) {
    float v;
    if (c < AD) {
      v = h[(size_t)bn * AD + c];
    } else if (c < AD + 32) {
      v = z_emb[(size_t)z[bn] * 32 + (c - AD)];
    } else if (c < AD + 64) {
      int j = c - (AD + 32);
      float center = RCUT * (float)j / 31.0f;
      float width = RCUT / 31.0f;
      float x = (fminf(r, RCUT) - center) / width;
      v = __expf(-0.5f * x * x);
    } else if (c < 195) {
      float u = (c == 192) ? px : ((c == 193) ? py : pz);
      v = u / rs;
    } else if (c == 195) {
      v = (n == ai) ? 1.f : 0.f;
    } else {
      v = 0.f;  // zero padding 196..223
    }
    atom[(size_t)bn * KATOM + c] = (_Float16)v;
  }
  if (c == 0) {
    float cc = 0.5f * (cosf(3.14159265358979323846f * r / RCUT) + 1.f) *
               ((r < RCUT) ? 1.f : 0.f);
    logcut[bn] = logf(fmaxf(cc, 1e-8f));
    validArr[bn] = (mask[bn] != 0 && r <= RCUT) ? 1 : 0;
  }
}

// ---------------------------------------------------------------------------
// Generic f16 WMMA GEMM: out[M,256] = act(A[M,K] @ W[K,256] + bias)
// W staged pre-transposed as Wt[256][K]. 128 threads = 4 waves, 64x64 tile.
// Double-buffered LDS, filled by async global->LDS DMA (no VGPR round-trip);
// one K-step prefetched ahead (steady state waits asynccnt<=4), last step
// peeled so the hot loop is branch-free.
// ---------------------------------------------------------------------------
__global__ __launch_bounds__(128) void gemm_f16_wmma_kernel(
    const _Float16* __restrict__ A, const _Float16* __restrict__ Wt,
    const float* __restrict__ bias, _Float16* __restrict__ out16,
    float* __restrict__ out32, int K, int act) {
  __shared__ alignas(16) _Float16 sA[2][64][32];
  __shared__ alignas(16) _Float16 sB[2][64][32];
  const int tid = threadIdx.x;
  const int wave = tid >> 5;
  const int lane = tid & 31;
  const int tm = blockIdx.x * 64;
  const int tn = blockIdx.y * 64;
  const int rt = tid >> 1;             // 0..63 : row within tile
  const int ct = (tid & 1) << 4;       // 0 or 16 halves
  v8f acc[4] = {};

  const _Float16* ga = A + (size_t)(tm + rt) * K + ct;   // bumped +32/stage
  const _Float16* gb = Wt + (size_t)(tn + rt) * K + ct;
  const unsigned laBase[2] = {lds_off(&sA[0][rt][ct]), lds_off(&sA[1][rt][ct])};
  const unsigned lbBase[2] = {lds_off(&sB[0][rt][ct]), lds_off(&sB[1][rt][ct])};

  auto issue_stage = [&](int buf, const _Float16* pa, const _Float16* pb) {
    async_g2l_b128(laBase[buf], pa);        // halves [ct .. ct+7]
    async_g2l_b128_o16(laBase[buf], pa);    // halves [ct+8 .. ct+15]
    async_g2l_b128(lbBase[buf], pb);
    async_g2l_b128_o16(lbBase[buf], pb);
  };
  auto compute_step = [&](int cur) {
    v16h af = frag_a_lds(&sA[cur][wave * 16][0], 32);
#pragma unroll
    for (int j = 0; j < 4; ++j) {
      v16h bf = frag_b_lds(&sB[cur][j * 16][0], 32);
      acc[j] = wmma_f16(af, bf, acc[j]);
    }
  };

  const int nsteps = K >> 5;
  issue_stage(0, ga, gb);
  ga += 32;
  gb += 32;
  for (int s = 0; s < nsteps - 1; ++s) {
    issue_stage(1 - (s & 1), ga, gb);   // prefetch stage s+1
    ga += 32;
    gb += 32;
    wait_async_le4();                   // stage s landed; s+1 in flight
    __syncthreads();                    // visible to all waves
    compute_step(s & 1);
    __syncthreads();                    // everyone done reading before refill
  }
  wait_async_0();                       // last stage landed
  __syncthreads();
  compute_step((nsteps - 1) & 1);

  const int nloc = lane & 15;
  const int moff = (lane & 16) ? 8 : 0;
#pragma unroll
  for (int j = 0; j < 4; ++j) {
    int n = tn + j * 16 + nloc;
    float bv = bias[n];
#pragma unroll
    for (int r = 0; r < 8; ++r) {
      int m = tm + wave * 16 + moff + r;
      float x = acc[j][r] + bv;
      if (act) x = silu_f(x);
      if (out16) out16[(size_t)m * 256 + n] = (_Float16)x;
      if (out32) out32[(size_t)m * 256 + n] = x;
    }
  }
}

// ---------------------------------------------------------------------------
// Attention: per (64-energy block, head, batch). QK^T (one WMMA K-step,
// K=HD=32), scale+radial-bias+mask, f32 softmax over 64 atoms, P@V via WMMA.
// Q/K tiles staged via async global->LDS DMA; V staged manually (transpose).
// ---------------------------------------------------------------------------
__global__ __launch_bounds__(128) void attention_kernel(
    const _Float16* __restrict__ Q, const _Float16* __restrict__ Kf,
    const _Float16* __restrict__ Vf, const float* __restrict__ logcut,
    const int* __restrict__ validArr, _Float16* __restrict__ AO) {
  __shared__ alignas(16) _Float16 sQ[64][32];
  __shared__ alignas(16) _Float16 sK[64][32];   // Bt for QK^T: rows = atoms
  __shared__ alignas(16) _Float16 sVt[32][64];  // Bt for P@V: [dim][atom]
  __shared__ alignas(16) float    sS[64][64];
  __shared__ alignas(16) _Float16 sP[64][64];
  __shared__ float sAdd[64];
  __shared__ float sAm[64];

  const int tid = threadIdx.x;
  const int wave = tid >> 5;
  const int lane = tid & 31;
  const int mblk = blockIdx.x;  // 0..3  (64 energies each)
  const int head = blockIdx.y;  // 0..7
  const int b = blockIdx.z;     // 0..15
  const int rt = tid >> 1;
  const int ct = (tid & 1) << 4;

  {
    const _Float16* gq =
        Q + (size_t)(b * NE + mblk * 64 + rt) * LATENT + head * HD + ct;
    const unsigned lq = lds_off(&sQ[rt][ct]);
    async_g2l_b128(lq, gq);
    async_g2l_b128_o16(lq, gq);
    const _Float16* gk =
        Kf + (size_t)(b * NATOM + rt) * LATENT + head * HD + ct;
    const unsigned lk = lds_off(&sK[rt][ct]);
    async_g2l_b128(lk, gk);
    async_g2l_b128_o16(lk, gk);
    const _Float16* gv = Vf + (size_t)(b * NATOM + rt) * LATENT + head * HD + ct;
    v8h v0 = *(const v8h*)gv;
    v8h v1 = *(const v8h*)(gv + 8);
#pragma unroll
    for (int i = 0; i < 8; ++i) {
      sVt[ct + i][rt] = v0[i];
      sVt[ct + 8 + i][rt] = v1[i];
    }
  }
  if (tid < 64) {
    int v = validArr[b * NATOM + tid];
    sAm[tid] = v ? 1.f : 0.f;
    sAdd[tid] = v ? logcut[b * NATOM + tid] : -1e9f;
  }
  wait_async_0();
  __syncthreads();

  {  // scores
    v16h qf = frag_a_lds(&sQ[wave * 16][0], 32);
    const int nloc = lane & 15;
    const int moff = (lane & 16) ? 8 : 0;
#pragma unroll
    for (int j = 0; j < 4; ++j) {
      v16h kf = frag_b_lds(&sK[j * 16][0], 32);
      v8f zero = {};
      v8f s = wmma_f16(qf, kf, zero);
      int n = j * 16 + nloc;
      float add = sAdd[n];
#pragma unroll
      for (int r = 0; r < 8; ++r)
        sS[wave * 16 + moff + r][n] = s[r] * 0.17677669529663689f + add;
    }
  }
  __syncthreads();

  if (tid < 64) {  // softmax over 64 atoms, one row per thread
    float mx = -3.0e38f;
    for (int n = 0; n < 64; ++n) mx = fmaxf(mx, sS[tid][n]);
    float sum = 0.f;
    for (int n = 0; n < 64; ++n) sum += __expf(sS[tid][n] - mx);
    float inv = 1.f / sum;
    for (int n = 0; n < 64; ++n)
      sP[tid][n] = (_Float16)(__expf(sS[tid][n] - mx) * inv * sAm[n]);
  }
  __syncthreads();

  {  // out = P @ V  (K = 64 atoms, N = 32 dims)
    v8f o[2] = {};
#pragma unroll
    for (int k0 = 0; k0 < 64; k0 += 32) {
      v16h pf = frag_a_lds(&sP[wave * 16][k0], 64);
#pragma unroll
      for (int j = 0; j < 2; ++j) {
        v16h vf = frag_b_lds(&sVt[j * 16][k0], 64);
        o[j] = wmma_f16(pf, vf, o[j]);
      }
    }
    const int nloc = lane & 15;
    const int moff = (lane & 16) ? 8 : 0;
#pragma unroll
    for (int j = 0; j < 2; ++j) {
#pragma unroll
      for (int r = 0; r < 8; ++r) {
        int m = b * NE + mblk * 64 + wave * 16 + moff + r;
        int d = head * HD + j * 16 + nloc;
        AO[(size_t)m * LATENT + d] = (_Float16)o[j][r];
      }
    }
  }
}

// ---------------------------------------------------------------------------
// Host orchestration
// ---------------------------------------------------------------------------
extern "C" void kernel_launch(void* const* d_in, const int* in_sizes, int n_in,
                              void* d_out, int out_size, void* d_ws,
                              size_t ws_size, hipStream_t stream) {
  (void)in_sizes; (void)n_in; (void)out_size; (void)ws_size;
  const float* h      = (const float*)d_in[0];
  const int*   z      = (const int*)d_in[1];
  const float* pos    = (const float*)d_in[2];
  const unsigned char* mask = (const unsigned char*)d_in[3];
  const float* e_feat = (const float*)d_in[4];
  const int*   absIdx = (const int*)d_in[5];
  // params pytree flattened alphabetically: k (3x(w,b)), o (2x), q (3x), v (3x), z_emb
  const float* kw[3] = {(const float*)d_in[6],  (const float*)d_in[8],  (const float*)d_in[10]};
  const float* kb[3] = {(const float*)d_in[7],  (const float*)d_in[9],  (const float*)d_in[11]};
  const float* ow[2] = {(const float*)d_in[12], (const float*)d_in[14]};
  const float* ob[2] = {(const float*)d_in[13], (const float*)d_in[15]};
  const float* qw[3] = {(const float*)d_in[16], (const float*)d_in[18], (const float*)d_in[20]};
  const float* qb[3] = {(const float*)d_in[17], (const float*)d_in[19], (const float*)d_in[21]};
  const float* vw[3] = {(const float*)d_in[22], (const float*)d_in[24], (const float*)d_in[26]};
  const float* vb[3] = {(const float*)d_in[23], (const float*)d_in[25], (const float*)d_in[27]};
  const float* z_emb = (const float*)d_in[28];

  char* ws = (char*)d_ws;
  size_t off = 0;
  auto alloc  = [&](size_t bytes) -> void* {
    void* p = ws + off;
    off = (off + bytes + 255) & ~(size_t)255;
    return p;
  };
  auto allocH = [&](size_t halves) -> _Float16* { return (_Float16*)alloc(halves * 2); };

  _Float16* wq0t = allocH((size_t)256 * KQ);
  _Float16* wq1t = allocH((size_t)256 * 256);
  _Float16* wq2t = allocH((size_t)256 * 256);
  _Float16* wk0t = allocH((size_t)256 * KATOM);
  _Float16* wk1t = allocH((size_t)256 * 256);
  _Float16* wk2t = allocH((size_t)256 * 256);
  _Float16* wv0t = allocH((size_t)256 * KATOM);
  _Float16* wv1t = allocH((size_t)256 * 256);
  _Float16* wv2t = allocH((size_t)256 * 256);
  _Float16* wo0t = allocH((size_t)256 * 256);
  _Float16* wo1t = allocH((size_t)256 * 256);

  _Float16* qin  = allocH((size_t)B_ * NE * KQ);
  _Float16* bufA = allocH((size_t)B_ * NE * 256);
  _Float16* bufB = allocH((size_t)B_ * NE * 256);
  _Float16* qfin = allocH((size_t)B_ * NE * 256);
  _Float16* atom = allocH((size_t)B_ * NATOM * KATOM);
  _Float16* kA   = allocH((size_t)B_ * NATOM * 256);
  _Float16* kB_  = allocH((size_t)B_ * NATOM * 256);
  _Float16* kfin = allocH((size_t)B_ * NATOM * 256);
  _Float16* vA   = allocH((size_t)B_ * NATOM * 256);
  _Float16* vB_  = allocH((size_t)B_ * NATOM * 256);
  _Float16* vfin = allocH((size_t)B_ * NATOM * 256);
  _Float16* ao   = allocH((size_t)B_ * NE * 256);
  _Float16* ohid = allocH((size_t)B_ * NE * 256);
  float* logcut  = (float*)alloc((size_t)B_ * NATOM * 4);
  int*   validA  = (int*)alloc((size_t)B_ * NATOM * 4);

  // 1) weight convert + transpose (+pad) to f16 Wt[N][Kpad]
  wt_cvt_kernel<<<(256 * KQ) / 256, 256, 0, stream>>>(qw[0], wq0t, KQ, KQ, 256);
  wt_cvt_kernel<<<256, 256, 0, stream>>>(qw[1], wq1t, 256, 256, 256);
  wt_cvt_kernel<<<256, 256, 0, stream>>>(qw[2], wq2t, 256, 256, 256);
  wt_cvt_kernel<<<(256 * KATOM) / 256, 256, 0, stream>>>(kw[0], wk0t, KATOM_R, KATOM, 256);
  wt_cvt_kernel<<<256, 256, 0, stream>>>(kw[1], wk1t, 256, 256, 256);
  wt_cvt_kernel<<<256, 256, 0, stream>>>(kw[2], wk2t, 256, 256, 256);
  wt_cvt_kernel<<<(256 * KATOM) / 256, 256, 0, stream>>>(vw[0], wv0t, KATOM_R, KATOM, 256);
  wt_cvt_kernel<<<256, 256, 0, stream>>>(vw[1], wv1t, 256, 256, 256);
  wt_cvt_kernel<<<256, 256, 0, stream>>>(vw[2], wv2t, 256, 256, 256);
  wt_cvt_kernel<<<256, 256, 0, stream>>>(ow[0], wo0t, 256, 256, 256);
  wt_cvt_kernel<<<256, 256, 0, stream>>>(ow[1], wo1t, 256, 256, 256);

  // 2) feature builders
  build_qin_kernel<<<(B_ * NE * KQ) / 256, 256, 0, stream>>>(h, e_feat, absIdx, qin);
  build_atom_kernel<<<B_ * NATOM, 256, 0, stream>>>(h, z, pos, mask, z_emb,
                                                    absIdx, atom, logcut, validA);

  // 3) q / k / v MLPs via WMMA GEMM
  dim3 gq(B_ * NE / 64, 4);
  gemm_f16_wmma_kernel<<<gq, 128, 0, stream>>>(qin,  wq0t, qb[0], bufA, nullptr, KQ,  1);
  gemm_f16_wmma_kernel<<<gq, 128, 0, stream>>>(bufA, wq1t, qb[1], bufB, nullptr, 256, 1);
  gemm_f16_wmma_kernel<<<gq, 128, 0, stream>>>(bufB, wq2t, qb[2], qfin, nullptr, 256, 0);

  dim3 gk(B_ * NATOM / 64, 4);
  gemm_f16_wmma_kernel<<<gk, 128, 0, stream>>>(atom, wk0t, kb[0], kA,   nullptr, KATOM, 1);
  gemm_f16_wmma_kernel<<<gk, 128, 0, stream>>>(kA,   wk1t, kb[1], kB_,  nullptr, 256,   1);
  gemm_f16_wmma_kernel<<<gk, 128, 0, stream>>>(kB_,  wk2t, kb[2], kfin, nullptr, 256,   0);
  gemm_f16_wmma_kernel<<<gk, 128, 0, stream>>>(atom, wv0t, vb[0], vA,   nullptr, KATOM, 1);
  gemm_f16_wmma_kernel<<<gk, 128, 0, stream>>>(vA,   wv1t, vb[1], vB_,  nullptr, 256,   1);
  gemm_f16_wmma_kernel<<<gk, 128, 0, stream>>>(vB_,  wv2t, vb[2], vfin, nullptr, 256,   0);

  // 4) attention
  dim3 ga(NE / 64, NH, B_);
  attention_kernel<<<ga, 128, 0, stream>>>(qfin, kfin, vfin, logcut, validA, ao);

  // 5) output MLP (final layer -> f32 d_out)
  gemm_f16_wmma_kernel<<<gq, 128, 0, stream>>>(ao,   wo0t, ob[0], ohid, nullptr,       256, 1);
  gemm_f16_wmma_kernel<<<gq, 128, 0, stream>>>(ohid, wo1t, ob[1], nullptr, (float*)d_out, 256, 0);
}
